// DiscreteAutoencoder_1288490188901
// MI455X (gfx1250) — matrix operand order repaired and checked by reading, not compiled, not run on hardware
//
#include <hip/hip_runtime.h>
#include <hip/hip_bf16.h>

typedef __bf16 bf16;
typedef __attribute__((ext_vector_type(16))) __bf16 v16bf;
typedef __attribute__((ext_vector_type(8)))  __bf16 v8bf;
typedef __attribute__((ext_vector_type(8)))  float  v8f;
typedef __attribute__((ext_vector_type(4)))  unsigned int v4u;
typedef __attribute__((ext_vector_type(8)))  int v8i;
typedef __attribute__((ext_vector_type(4)))  int v4i;

#define B_ROWS   32768
#define S_DIM    256
#define L_DIM    64
#define H_DIM    64
#define N_EMB    2048
#define ROWS_PER_BLOCK 128

// ws layout: [0, 256KB) bf16 codebook (row-major, unpadded), then 2048 f32 norms
#define WS_EMB_BYTES   ((size_t)N_EMB * L_DIM * 2)
#define WS_NORM_OFF    WS_EMB_BYTES
#define WS_NEEDED      (WS_EMB_BYTES + (size_t)N_EMB * 4)

// D = A*B + C, A 16x32 bf16, B 32x16 bf16, C/D 16x16 f32
__device__ __forceinline__ v8f wmma_bf16(v16bf a, v16bf b, v8f c) {
  return __builtin_amdgcn_wmma_f32_16x16x32_bf16(
      false, a, false, b, (short)0, c, false, false);
}

// A-fragment (16x32 bf16) for lane (m = lane&15, h = lane>>4) from a
// row-major f32 row pointer (already offset to A[m] + k_base), scaled by s.
__device__ __forceinline__ v16bf make_afrag(const float* rp, int h, float s) {
  v16bf a;
#pragma unroll
  for (int j = 0; j < 8; ++j) a[j]     = (bf16)(s * rp[h * 8 + j]);
#pragma unroll
  for (int j = 0; j < 8; ++j) a[8 + j] = (bf16)(s * rp[16 + h * 8 + j]);
  return a;
}

// B-fragment (32x16 bf16): elems[j] = B[k_base + h*8 + j][n], elems[8+j] = B[k_base + 16 + h*8 + j][n].
__device__ __forceinline__ v16bf make_bfrag(const bf16* colptr, int h) {
  union { v16bf v; v8bf q[2]; } u;
  u.q[0] = *(const v8bf*)(colptr + h * 8);
  u.q[1] = *(const v8bf*)(colptr + 16 + h * 8);
  return u.v;
}

__device__ __forceinline__ v8f bcast8(float v) {
  v8f c;
#pragma unroll
  for (int r = 0; r < 8; ++r) c[r] = v;
  return c;
}

// ---------------- Kernel 0: codebook preconvert (one-shot, tiny) -------------
__global__ __launch_bounds__(256) void vq_precompute_kernel(
    const float* __restrict__ emb, bf16* __restrict__ emb_bf,
    float* __restrict__ norms_g) {
  const int e = blockIdx.x * 256 + threadIdx.x;   // 2048 rows, 8 blocks
  if (e >= N_EMB) return;
  const float* er = emb + (size_t)e * L_DIM;
  bf16* wr = emb_bf + (size_t)e * L_DIM;
  float ss = 0.0f;
#pragma unroll 8
  for (int j = 0; j < L_DIM; ++j) {
    float v = er[j];
    ss += v * v;
    wr[j] = (bf16)v;
  }
  norms_g[e] = ss;
}

// ---------------- Kernel 1: encoder  z_e = relu(x@W1+b1)@W2+b2 ----------------
__global__ __launch_bounds__(256) void vq_encoder_kernel(
    const float* __restrict__ x,  const float* __restrict__ w1,
    const float* __restrict__ b1, const float* __restrict__ w2,
    const float* __restrict__ b2, float* __restrict__ z_e) {
  __shared__ bf16  w1c[64][264];        // w1c[n][k] = W1[k][n], 16B-aligned rows
  __shared__ bf16  w2c[64][72];
  __shared__ float b1s[64], b2s[64];
  __shared__ float hbuf[8][16][68];     // per-wave H tile (f32), padded rows

  const int t = threadIdx.x;
  for (int i = t; i < 256 * 64; i += 256) w1c[i & 63][i >> 6] = (bf16)w1[i];
  for (int i = t; i < 64 * 64;  i += 256) w2c[i & 63][i >> 6] = (bf16)w2[i];
  if (t < 64) { b1s[t] = b1[t]; b2s[t] = b2[t]; }
  __syncthreads();

  const int wave = t >> 5, lane = t & 31;
  const int m = lane & 15, h = lane >> 4, q = m;
  const int rowbase = blockIdx.x * ROWS_PER_BLOCK + wave * 16;

  // ---- GEMM1: [16,256] x [256,64] ----
  v16bf afr[8];
  const float* xrow = x + (size_t)(rowbase + m) * S_DIM;
#pragma unroll
  for (int kc = 0; kc < 8; ++kc) afr[kc] = make_afrag(xrow + kc * 32, h, 1.0f);

#pragma unroll
  for (int nt = 0; nt < 4; ++nt) {
    v8f c = bcast8(b1s[nt * 16 + q]);
#pragma unroll
    for (int kc = 0; kc < 8; ++kc)
      c = wmma_bf16(afr[kc], make_bfrag(&w1c[nt * 16 + q][kc * 32], h), c);
#pragma unroll
    for (int r = 0; r < 8; ++r)
      hbuf[wave][r + 8 * h][nt * 16 + q] = fmaxf(c[r], 0.0f);
  }
  __syncthreads();

  // ---- GEMM2: [16,64] x [64,64] ----
  v16bf a2[2];
#pragma unroll
  for (int kc = 0; kc < 2; ++kc)
    a2[kc] = make_afrag(&hbuf[wave][m][kc * 32], h, 1.0f);

#pragma unroll
  for (int nt = 0; nt < 4; ++nt) {
    v8f c = bcast8(b2s[nt * 16 + q]);
#pragma unroll
    for (int kc = 0; kc < 2; ++kc)
      c = wmma_bf16(a2[kc], make_bfrag(&w2c[nt * 16 + q][kc * 32], h), c);
#pragma unroll
    for (int r = 0; r < 8; ++r)
      z_e[(size_t)(rowbase + r + 8 * h) * L_DIM + nt * 16 + q] = c[r];
  }
}

// ---------------- shared VQ core: argmin_n(||e_n||^2 - 2 z.e_n) + gather -----
__device__ __forceinline__ void vq_core(
    const float* __restrict__ z_e, const float* __restrict__ emb,
    float* __restrict__ z_q, const bf16 (*embs)[72], const float* norms) {
  const int t = threadIdx.x;
  const int wave = t >> 5, lane = t & 31;
  const int m = lane & 15, h = lane >> 4, q = m;
  const int rowbase = blockIdx.x * ROWS_PER_BLOCK + wave * 16;

  // A = -2 * z_e tile (D = ||e||^2 - 2 z.e with C = ||e||^2)
  const float* zr = z_e + (size_t)(rowbase + m) * L_DIM;
  v16bf a0 = make_afrag(zr,      h, -2.0f);
  v16bf a1 = make_afrag(zr + 32, h, -2.0f);

  float minv[8];
  int   mini[8];
#pragma unroll
  for (int r = 0; r < 8; ++r) { minv[r] = 3.4e38f; mini[r] = 0; }

#pragma unroll 4
  for (int nt = 0; nt < N_EMB / 16; ++nt) {
    const int n = nt * 16 + q;                // this lane's code column
    v8f c = bcast8(norms[n]);
    c = wmma_bf16(a0, make_bfrag(&embs[n][0],  h), c);
    c = wmma_bf16(a1, make_bfrag(&embs[n][32], h), c);
#pragma unroll
    for (int r = 0; r < 8; ++r) {
      if (c[r] < minv[r]) { minv[r] = c[r]; mini[r] = n; }
    }
  }

  // reduce across the 16 lanes of each half (wave32: halves stay disjoint)
#pragma unroll
  for (int off = 1; off < 16; off <<= 1) {
#pragma unroll
    for (int r = 0; r < 8; ++r) {
      float ov = __shfl_xor(minv[r], off, 32);
      int   oi = __shfl_xor(mini[r], off, 32);
      if (ov < minv[r] || (ov == minv[r] && oi < mini[r])) {
        minv[r] = ov; mini[r] = oi;
      }
    }
  }

  // gather z_q rows from the f32 codebook (full precision output)
#pragma unroll
  for (int r = 0; r < 8; ++r) {
    const int grow = rowbase + r + 8 * h;
    const float4 v = *(const float4*)(emb + (size_t)mini[r] * L_DIM + q * 4);
    *(float4*)(z_q + (size_t)grow * L_DIM + q * 4) = v;
  }
}

// ---------------- Kernel 2a: VQ with TDM codebook staging --------------------
__global__ __launch_bounds__(256) void vq_quantize_tdm_kernel(
    const float* __restrict__ z_e, const float* __restrict__ emb,
    const bf16* __restrict__ emb_bf, const float* __restrict__ norms_g,
    float* __restrict__ z_q) {
  __shared__ bf16  embs[N_EMB][72];     // 2048 x (128B data + 16B pad) = 288KB
  __shared__ float norms[N_EMB];

  const int t = threadIdx.x;
  // norms: plain coalesced global->LDS
  for (int e = t; e < N_EMB; e += 256) norms[e] = norms_g[e];

  if (t < 32) {
    // One TDM op DMAs the whole bf16 codebook into LDS, inserting the 16B
    // row pad via D# pad fields (pad_interval=4 -> every 32 DWORDs,
    // pad_amount=3 -> 4 DWORDs). Tracked on TENSORcnt.
    const unsigned long long ga = (unsigned long long)(uintptr_t)emb_bf;
    const unsigned int lds_off = (unsigned int)(uintptr_t)&embs[0][0];
    v4u g0 = { 1u,                                    // count=1, user mode
               lds_off,                               // lds_addr
               (unsigned int)(ga & 0xFFFFFFFFu),      // global_addr[31:0]
               (unsigned int)((ga >> 32) & 0x01FFFFFFu) | (2u << 30) }; // type=2
    v8i g1 = { (int)((1u << 16) |                     // data_size = 2 bytes
                     (1u << 20) |                     // pad_enable
                     (4u << 22) |                     // pad_interval: 32 DWORDs
                     (3u << 25)),                     // pad_amount: 4 DWORDs
               (int)(64u << 16),                      // tensor_dim0 = 64 (lo16)
               (int)(2048u << 16),                    // dim0 hi=0 | tensor_dim1 lo16
               (int)(64u << 16),                      // dim1 hi=0 | tile_dim0 = 64
               2048,                                  // tile_dim1 | tile_dim2=0
               64,                                    // tensor_dim0_stride lo32
               0, 0 };
    v4i g2 = { 0, 0, 0, 0 };
    v4i g3 = { 0, 0, 0, 0 };
#if __clang_major__ >= 23
    v8i g4 = { 0, 0, 0, 0, 0, 0, 0, 0 };
    __builtin_amdgcn_tensor_load_to_lds(g0, g1, g2, g3, g4, 0);
#else
    __builtin_amdgcn_tensor_load_to_lds(g0, g1, g2, g3, 0);
#endif
    __builtin_amdgcn_s_wait_tensorcnt(0);
  }
  __syncthreads();

  vq_core(z_e, emb, z_q, embs, norms);
}

// ---------------- Kernel 2b: VQ with in-kernel staging (ws fallback) ---------
__global__ __launch_bounds__(256) void vq_quantize_kernel(
    const float* __restrict__ z_e, const float* __restrict__ emb,
    float* __restrict__ z_q) {
  __shared__ bf16  embs[N_EMB][72];
  __shared__ float norms[N_EMB];

  const int t = threadIdx.x;
  for (int e = t; e < N_EMB; e += 256) {
    const float* er = emb + (size_t)e * L_DIM;
    float ss = 0.0f;
#pragma unroll 4
    for (int j = 0; j < L_DIM; ++j) {
      float v = er[j];
      ss += v * v;
      embs[e][j] = (bf16)v;
    }
    norms[e] = ss;
  }
  __syncthreads();

  vq_core(z_e, emb, z_q, embs, norms);
}

// ---------------- Kernel 3: decoder  x_recon = relu(z_q@D1+b1)@D2+b2 ---------
__global__ __launch_bounds__(256) void vq_decoder_kernel(
    const float* __restrict__ z_q, const float* __restrict__ w1,
    const float* __restrict__ b1,  const float* __restrict__ w2,
    const float* __restrict__ b2,  float* __restrict__ recon) {
  __shared__ bf16  d1c[64][72];         // d1c[n][k] = D1[k][n]
  __shared__ bf16  d2c[256][72];        // d2c[n][k] = D2[k][n]
  __shared__ float b1s[64];
  __shared__ float b2s[256];
  __shared__ float hbuf[8][16][68];

  const int t = threadIdx.x;
  for (int i = t; i < 64 * 64;  i += 256) d1c[i & 63][i >> 6]  = (bf16)w1[i];
  for (int i = t; i < 64 * 256; i += 256) d2c[i & 255][i >> 8] = (bf16)w2[i];
  if (t < 64) b1s[t] = b1[t];
  b2s[t] = b2[t];
  __syncthreads();

  const int wave = t >> 5, lane = t & 31;
  const int m = lane & 15, h = lane >> 4, q = m;
  const int rowbase = blockIdx.x * ROWS_PER_BLOCK + wave * 16;

  // ---- GEMM1: [16,64] x [64,64] ----
  const float* zr = z_q + (size_t)(rowbase + m) * L_DIM;
  v16bf a0 = make_afrag(zr,      h, 1.0f);
  v16bf a1 = make_afrag(zr + 32, h, 1.0f);

#pragma unroll
  for (int nt = 0; nt < 4; ++nt) {
    v8f c = bcast8(b1s[nt * 16 + q]);
    c = wmma_bf16(a0, make_bfrag(&d1c[nt * 16 + q][0],  h), c);
    c = wmma_bf16(a1, make_bfrag(&d1c[nt * 16 + q][32], h), c);
#pragma unroll
    for (int r = 0; r < 8; ++r)
      hbuf[wave][r + 8 * h][nt * 16 + q] = fmaxf(c[r], 0.0f);
  }
  __syncthreads();

  // ---- GEMM2: [16,64] x [64,256] ----
  v16bf h0 = make_afrag(&hbuf[wave][m][0],  h, 1.0f);
  v16bf h1 = make_afrag(&hbuf[wave][m][32], h, 1.0f);

#pragma unroll
  for (int nt = 0; nt < 16; ++nt) {
    v8f c = bcast8(b2s[nt * 16 + q]);
    c = wmma_bf16(h0, make_bfrag(&d2c[nt * 16 + q][0],  h), c);
    c = wmma_bf16(h1, make_bfrag(&d2c[nt * 16 + q][32], h), c);
#pragma unroll
    for (int r = 0; r < 8; ++r)
      recon[(size_t)(rowbase + r + 8 * h) * S_DIM + nt * 16 + q] = c[r];
  }
}

extern "C" void kernel_launch(void* const* d_in, const int* in_sizes, int n_in,
                              void* d_out, int out_size, void* d_ws, size_t ws_size,
                              hipStream_t stream) {
  const float* x      = (const float*)d_in[0];
  const float* enc_w1 = (const float*)d_in[1];
  const float* enc_b1 = (const float*)d_in[2];
  const float* enc_w2 = (const float*)d_in[3];
  const float* enc_b2 = (const float*)d_in[4];
  const float* emb    = (const float*)d_in[5];
  const float* dec_w1 = (const float*)d_in[6];
  const float* dec_b1 = (const float*)d_in[7];
  const float* dec_w2 = (const float*)d_in[8];
  const float* dec_b2 = (const float*)d_in[9];

  float* out   = (float*)d_out;
  float* recon = out;                                    // [32768, 256]
  float* z_e   = out + (size_t)B_ROWS * S_DIM;           // [32768, 64]
  float* z_q   = z_e + (size_t)B_ROWS * L_DIM;           // [32768, 64]

  const int blocks = B_ROWS / ROWS_PER_BLOCK;            // 256

  vq_encoder_kernel<<<blocks, 256, 0, stream>>>(x, enc_w1, enc_b1, enc_w2,
                                                enc_b2, z_e);

  if (ws_size >= WS_NEEDED) {
    bf16*  emb_bf  = (bf16*)d_ws;
    float* norms_g = (float*)((char*)d_ws + WS_NORM_OFF);
    vq_precompute_kernel<<<N_EMB / 256, 256, 0, stream>>>(emb, emb_bf, norms_g);
    vq_quantize_tdm_kernel<<<blocks, 256, 0, stream>>>(z_e, emb, emb_bf,
                                                       norms_g, z_q);
  } else {
    vq_quantize_kernel<<<blocks, 256, 0, stream>>>(z_e, emb, z_q);
  }

  vq_decoder_kernel<<<blocks, 256, 0, stream>>>(z_q, dec_w1, dec_b1, dec_w2,
                                                dec_b2, recon);
}
